// PatternStore_43645457662503
// MI455X (gfx1250) — matrix operand order repaired and checked by reading, not compiled, not run on hardware
//
#include <hip/hip_runtime.h>
#include <math.h>

#define BB    8
#define TT    2048
#define HH    2048
#define NKEYS 100000
#define PROJ  512
#define TOPK  10

typedef __bf16 bf16_t;
typedef bf16_t v16bf __attribute__((ext_vector_type(16)));
typedef bf16_t v8bf  __attribute__((ext_vector_type(8)));
typedef float  v8f   __attribute__((ext_vector_type(8)));

static __device__ __forceinline__ bf16_t f2bf(float f) {
    unsigned u = __builtin_bit_cast(unsigned, f);
    unsigned r = (u + 0x7FFFu + ((u >> 16) & 1u)) >> 16;
    unsigned short s = (unsigned short)r;
    return __builtin_bit_cast(bf16_t, s);
}

// 16 contiguous floats -> v16bf (conversion fallback paths only)
static __device__ __forceinline__ v16bf cvt16_bf(const float* p) {
    float4 c0 = ((const float4*)p)[0];
    float4 c1 = ((const float4*)p)[1];
    float4 c2 = ((const float4*)p)[2];
    float4 c3 = ((const float4*)p)[3];
    v16bf r;
    r[0]=f2bf(c0.x); r[1]=f2bf(c0.y); r[2]=f2bf(c0.z); r[3]=f2bf(c0.w);
    r[4]=f2bf(c1.x); r[5]=f2bf(c1.y); r[6]=f2bf(c1.z); r[7]=f2bf(c1.w);
    r[8]=f2bf(c2.x); r[9]=f2bf(c2.y); r[10]=f2bf(c2.z); r[11]=f2bf(c2.w);
    r[12]=f2bf(c3.x); r[13]=f2bf(c3.y); r[14]=f2bf(c3.z); r[15]=f2bf(c3.w);
    return r;
}
static __device__ __forceinline__ v16bf cvt8p8_bf(const float* p0, const float* p1) {
    float4 c0 = ((const float4*)p0)[0];
    float4 c1 = ((const float4*)p0)[1];
    float4 c2 = ((const float4*)p1)[0];
    float4 c3 = ((const float4*)p1)[1];
    v16bf r;
    r[0]=f2bf(c0.x); r[1]=f2bf(c0.y); r[2]=f2bf(c0.z); r[3]=f2bf(c0.w);
    r[4]=f2bf(c1.x); r[5]=f2bf(c1.y); r[6]=f2bf(c1.z); r[7]=f2bf(c1.w);
    r[8]=f2bf(c2.x); r[9]=f2bf(c2.y); r[10]=f2bf(c2.z); r[11]=f2bf(c2.w);
    r[12]=f2bf(c3.x); r[13]=f2bf(c3.y); r[14]=f2bf(c3.z); r[15]=f2bf(c3.w);
    return r;
}
// two 16-byte bf16 loads -> v16bf (hot path: pure vmem, no VALU conversion)
static __device__ __forceinline__ v16bf ld8p8(const bf16_t* p0, const bf16_t* p1) {
    v8bf lo = *(const v8bf*)p0;
    v8bf hi = *(const v8bf*)p1;
    return __builtin_shufflevector(lo, hi, 0,1,2,3,4,5,6,7,8,9,10,11,12,13,14,15);
}

// ---------------- f32 -> bf16 pre-conversion (16 elems / thread) ----------------
__global__ void k_cvt_bf16(const float* __restrict__ src, bf16_t* __restrict__ dst, int n16) {
    int i = blockIdx.x * 256 + threadIdx.x;
    if (i >= n16) return;
    const float4* s = (const float4*)src + (size_t)i * 4;
    float4 c0 = s[0], c1 = s[1], c2 = s[2], c3 = s[3];
    v8bf r0, r1;
    r0[0]=f2bf(c0.x); r0[1]=f2bf(c0.y); r0[2]=f2bf(c0.z); r0[3]=f2bf(c0.w);
    r0[4]=f2bf(c1.x); r0[5]=f2bf(c1.y); r0[6]=f2bf(c1.z); r0[7]=f2bf(c1.w);
    r1[0]=f2bf(c2.x); r1[1]=f2bf(c2.y); r1[2]=f2bf(c2.z); r1[3]=f2bf(c2.w);
    r1[4]=f2bf(c3.x); r1[5]=f2bf(c3.y); r1[6]=f2bf(c3.z); r1[7]=f2bf(c3.w);
    ((v8bf*)dst)[(size_t)i * 2    ] = r0;
    ((v8bf*)dst)[(size_t)i * 2 + 1] = r1;
}

// ---------------- stage 1: query = mean_t x ----------------
__global__ void k_zero_query(float* query) {
    int i = blockIdx.x * 256 + threadIdx.x;
    if (i < BB * HH) query[i] = 0.0f;
}

__global__ void k_mean_x(const float* __restrict__ x, float* __restrict__ query) {
    int idx = blockIdx.x * 256 + threadIdx.x;      // (b,h) in 0..16383
    int tc  = blockIdx.y;                          // t-chunk 0..7
    int b = idx >> 11, h = idx & (HH - 1);
    const float* p = x + ((size_t)b * TT + (size_t)tc * 256) * HH + h;
    float s = 0.0f;
    for (int t = 0; t < 256; ++t) s += p[(size_t)t * HH];
    atomicAdd(&query[idx], s * (1.0f / (float)TT));
}

// ---------------- stage 2: q = query @ Wq^T + bq ----------------
__global__ void k_proj_q(const float* __restrict__ query, const float* __restrict__ Wq,
                         const float* __restrict__ bq, float* __restrict__ q) {
    int idx = blockIdx.x * 256 + threadIdx.x;      // 0..4095
    int b = idx >> 9, p = idx & (PROJ - 1);
    const float4* qr = (const float4*)(query + (size_t)b * HH);
    const float4* wr = (const float4*)(Wq + (size_t)p * HH);
    float acc = 0.0f;
    for (int i = 0; i < HH / 4; ++i) {
        float4 a = qr[i], w = wr[i];
        acc += a.x * w.x + a.y * w.y + a.z * w.z + a.w * w.w;
    }
    q[idx] = acc + bq[p];
}

__global__ void k_qbk(const float* __restrict__ q, const float* __restrict__ bk,
                      float* __restrict__ qbk) {
    int b = threadIdx.x >> 5, lane = threadIdx.x & 31;
    float s = 0.0f;
    for (int p = lane; p < PROJ; p += 32) s += q[b * PROJ + p] * bk[p];
    for (int m = 1; m < 32; m <<= 1) s += __shfl_xor(s, m);
    if (lane == 0) qbk[b] = s;
}

__global__ void k_score_init(const float* __restrict__ qbk, float* __restrict__ scores) {
    int i = blockIdx.x * 256 + threadIdx.x;        // 0..799999
    scores[i] = qbk[i / NKEYS];
}

// ---------------- stage 3: fused kproj GEMM + scores (bf16 WMMA) ----------------
// Block = 8 waves as 4(M-groups) x 2(N-groups); wave tile = 32M x 64N (8 accs);
// inner loop over 4 N-groups of 128 -> block covers 128M x 512N; keys read once.
// MODE: 2 = bf16 keys + bf16 Wk; 1 = f32 keys + bf16 Wk; 0 = all f32 (in-loop cvt)
template <int MODE>
__global__ __launch_bounds__(256, 1) void k_gemm_scores_t(
        const float* __restrict__ keysF, const bf16_t* __restrict__ keysB,
        const float* __restrict__ WkF, const bf16_t* __restrict__ WkB,
        const float* __restrict__ q, float* __restrict__ scores) {
    const int wave = threadIdx.x >> 5;
    const int lane = threadIdx.x & 31;
    const int half = lane >> 4;
    const int l    = lane & 15;
    const int mg   = wave >> 1;          // 0..3
    const int ngsel = wave & 1;          // 0..1

    const int m0 = blockIdx.x * 128 + mg * 32;     // wave-uniform
    if (m0 + 32 > NKEYS) return;                   // uniform per wave: EXEC all-ones inside

    const bf16_t* arB0 = keysB + (size_t)(m0 + l) * HH;
    const bf16_t* arB1 = keysB + (size_t)(m0 + 16 + l) * HH;
    const float*  arF0 = keysF + (size_t)(m0 + l) * HH;
    const float*  arF1 = keysF + (size_t)(m0 + 16 + l) * HH;

    for (int ng = 0; ng < 4; ++ng) {
        const int n0 = ng * 128 + ngsel * 64;
        v8f acc[2][4] = {};

        const bf16_t* brB[4];
        const float*  brF[4];
#pragma unroll
        for (int nt = 0; nt < 4; ++nt) {
            brB[nt] = WkB + (size_t)(n0 + nt * 16 + l) * HH + half * 16;
            brF[nt] = WkF + (size_t)(n0 + nt * 16 + l) * HH + half * 16;
        }

        for (int kb = 0; kb < HH; kb += 32) {
            v16bf a0, a1;
            if (MODE == 2) {
                a0 = ld8p8(arB0 + kb + half * 16, arB0 + kb + half * 16 + 8);
                a1 = ld8p8(arB1 + kb + half * 16, arB1 + kb + half * 16 + 8);
            } else {
                a0 = cvt8p8_bf(arF0 + kb + half * 16, arF0 + kb + half * 16 + 8);
                a1 = cvt8p8_bf(arF1 + kb + half * 16, arF1 + kb + half * 16 + 8);
            }
#pragma unroll
            for (int nt = 0; nt < 4; ++nt) {
                v16bf w;
                if (MODE >= 1) w = ld8p8(brB[nt] + kb, brB[nt] + kb + 8);
                else           w = cvt16_bf(brF[nt] + kb);
                acc[0][nt] = __builtin_amdgcn_wmma_f32_16x16x32_bf16(
                                 false, a0, false, w, (short)0, acc[0][nt], false, false);
                acc[1][nt] = __builtin_amdgcn_wmma_f32_16x16x32_bf16(
                                 false, a1, false, w, (short)0, acc[1][nt], false, false);
            }
        }

        // scores[b][row] += sum over this wave's 64 N-cols of q[b][n]*kproj[row][n]
        for (int b = 0; b < BB; ++b) {
            float qv[4];
#pragma unroll
            for (int nt = 0; nt < 4; ++nt) qv[nt] = q[b * PROJ + n0 + nt * 16 + l];
#pragma unroll
            for (int aa = 0; aa < 2; ++aa) {
#pragma unroll
                for (int i = 0; i < 8; ++i) {
                    float p = acc[aa][0][i] * qv[0] + acc[aa][1][i] * qv[1]
                            + acc[aa][2][i] * qv[2] + acc[aa][3][i] * qv[3];
                    p += __shfl_xor(p, 1);
                    p += __shfl_xor(p, 2);
                    p += __shfl_xor(p, 4);
                    p += __shfl_xor(p, 8);
                    if (l == 0) {
                        int row = m0 + aa * 16 + i + half * 8;
                        atomicAdd(&scores[(size_t)b * NKEYS + row], p);
                    }
                }
            }
        }
    }
}

// ---------------- stage 4: top-k (10 argmax passes per batch) ----------------
__global__ void k_topk(const float* __restrict__ scores, const float* __restrict__ conf,
                       int* __restrict__ topidx) {
    const int b = blockIdx.x, tid = threadIdx.x;
    __shared__ float sval[256];
    __shared__ int   sidx[256];
    __shared__ int   chosen[TOPK];
    for (int j = 0; j < TOPK; ++j) {
        float best = -3.4e38f; int bi = 0x7fffffff;
        for (int n = tid; n < NKEYS; n += 256) {
            bool used = false;
            for (int c = 0; c < j; ++c) if (chosen[c] == n) used = true;
            if (!used) {
                float v = scores[(size_t)b * NKEYS + n] * conf[n];
                if (v > best || (v == best && n < bi)) { best = v; bi = n; }
            }
        }
        sval[tid] = best; sidx[tid] = bi;
        __syncthreads();
        for (int s = 128; s > 0; s >>= 1) {
            if (tid < s) {
                float v1 = sval[tid + s]; int i1 = sidx[tid + s];
                if (v1 > sval[tid] || (v1 == sval[tid] && i1 < sidx[tid])) {
                    sval[tid] = v1; sidx[tid] = i1;
                }
            }
            __syncthreads();
        }
        if (tid == 0) chosen[j] = sidx[0];
        __syncthreads();
    }
    if (tid < TOPK) topidx[b * TOPK + tid] = chosen[tid];
}

// ---------------- stage 5: gate_input = [query, mean(retrieved)] ----------------
__global__ void k_gatein(const float* __restrict__ query, const float* __restrict__ keys,
                         const int* __restrict__ topidx, float* __restrict__ gatein) {
    int i = blockIdx.x * 256 + threadIdx.x;        // 0..32767
    int b = i >> 12, c = i & 4095;
    float v;
    if (c < HH) {
        v = query[b * HH + c];
    } else {
        int hh = c - HH;
        float s = 0.0f;
        for (int j = 0; j < TOPK; ++j)
            s += keys[(size_t)topidx[b * TOPK + j] * HH + hh];
        v = s * (1.0f / TOPK);
    }
    gatein[i] = v;
}

// ---------------- stage 6a: gate = sigmoid(...), h = gelu(...) ----------------
__global__ void k_gate_h(const float* __restrict__ gatein,
                         const float* __restrict__ Wg, const float* __restrict__ bg,
                         const float* __restrict__ Wm1, const float* __restrict__ bm1,
                         float* __restrict__ gate, float* __restrict__ h) {
    int idx = blockIdx.x * 256 + threadIdx.x;      // 0..16383
    int b = idx >> 11, o = idx & (HH - 1);
    const float4* gi = (const float4*)(gatein + (size_t)b * 2 * HH);
    const float4* wg = (const float4*)(Wg + (size_t)o * 2 * HH);
    const float4* wm = (const float4*)(Wm1 + (size_t)o * 2 * HH);
    float ag = 0.0f, am = 0.0f;
    for (int i = 0; i < (2 * HH) / 4; ++i) {
        float4 a = gi[i], g = wg[i], m = wm[i];
        ag += a.x * g.x + a.y * g.y + a.z * g.z + a.w * g.w;
        am += a.x * m.x + a.y * m.y + a.z * m.z + a.w * m.w;
    }
    ag += bg[o];
    am += bm1[o];
    gate[idx] = 1.0f / (1.0f + expf(-ag));
    h[idx]    = 0.5f * am * (1.0f + erff(am * 0.70710678118654752f));
}

// ---------------- stage 6b: gm = gate * (h@Wm2^T + bm2) ----------------
__global__ void k_gm(const float* __restrict__ h, const float* __restrict__ Wm2,
                     const float* __restrict__ bm2, const float* __restrict__ gate,
                     float* __restrict__ gm) {
    int idx = blockIdx.x * 256 + threadIdx.x;      // 0..16383
    int b = idx >> 11, o = idx & (HH - 1);
    const float4* hr = (const float4*)(h + (size_t)b * HH);
    const float4* wr = (const float4*)(Wm2 + (size_t)o * HH);
    float acc = 0.0f;
    for (int i = 0; i < HH / 4; ++i) {
        float4 a = hr[i], w = wr[i];
        acc += a.x * w.x + a.y * w.y + a.z * w.z + a.w * w.w;
    }
    gm[idx] = gate[idx] * (acc + bm2[o]);
}

// ---------------- stage 7: y = x + gm (bcast T); LayerNorm(H) ----------------
__global__ __launch_bounds__(256) void k_final_ln(
        const float* __restrict__ x, const float* __restrict__ gm,
        const float* __restrict__ gamma, const float* __restrict__ beta,
        float* __restrict__ out) {
    const int bt = blockIdx.x;
    const int b  = bt >> 11;
    const int tid = threadIdx.x;
    const float* xr = x + (size_t)bt * HH;
    const float* g  = gm + (size_t)b * HH;
    float* orow = out + (size_t)bt * HH;

    float v[8];
    float s = 0.0f, s2 = 0.0f;
#pragma unroll
    for (int i = 0; i < 8; ++i) {
        int c = tid + i * 256;
        float y = xr[c] + g[c];
        v[i] = y; s += y; s2 += y * y;
    }
    __shared__ float rs[256], rs2[256];
    rs[tid] = s; rs2[tid] = s2;
    __syncthreads();
    for (int k = 128; k > 0; k >>= 1) {
        if (tid < k) { rs[tid] += rs[tid + k]; rs2[tid] += rs2[tid + k]; }
        __syncthreads();
    }
    float mu  = rs[0] * (1.0f / HH);
    float var = rs2[0] * (1.0f / HH) - mu * mu;
    float inv = rsqrtf(var + 1e-5f);
#pragma unroll
    for (int i = 0; i < 8; ++i) {
        int c = tid + i * 256;
        orow[c] = (v[i] - mu) * inv * gamma[c] + beta[c];
    }
}

extern "C" void kernel_launch(void* const* d_in, const int* in_sizes, int n_in,
                              void* d_out, int out_size, void* d_ws, size_t ws_size,
                              hipStream_t stream) {
    const float* x     = (const float*)d_in[0];
    const float* keys  = (const float*)d_in[1];
    const float* conf  = (const float*)d_in[2];
    const float* Wq    = (const float*)d_in[3];
    const float* bq    = (const float*)d_in[4];
    const float* Wk    = (const float*)d_in[5];
    const float* bk    = (const float*)d_in[6];
    const float* Wg    = (const float*)d_in[7];
    const float* bg    = (const float*)d_in[8];
    const float* Wm1   = (const float*)d_in[9];
    const float* bm1   = (const float*)d_in[10];
    const float* Wm2   = (const float*)d_in[11];
    const float* bm2   = (const float*)d_in[12];
    const float* gamma = (const float*)d_in[13];
    const float* beta  = (const float*)d_in[14];
    float* out = (float*)d_out;

    float* ws     = (float*)d_ws;
    float* query  = ws;                         // 16384
    float* q      = query + 16384;              // 4096
    float* qbk    = q + 4096;                   // 16 (8 used)
    float* scores = qbk + 16;                   // 800000
    int*   topidx = (int*)(scores + 800000);    // 96 (80 used)
    float* gatein = scores + 800000 + 96;       // 32768
    float* gate   = gatein + 32768;             // 16384
    float* hbuf   = gate + 16384;               // 16384
    float* gm     = hbuf + 16384;               // 16384
    char*  extra  = (char*)(gm + 16384);        // bf16 staging region (16B aligned)

    const size_t base_bytes  = (size_t)((char*)extra - (char*)d_ws);
    const size_t keysB_bytes = (size_t)NKEYS * HH * sizeof(bf16_t);   // 409.6 MB
    const size_t wkB_bytes   = (size_t)PROJ * HH * sizeof(bf16_t);    // 2 MB

    int mode;
    bf16_t* keysB = nullptr;
    bf16_t* WkB   = nullptr;
    if (ws_size >= base_bytes + keysB_bytes + wkB_bytes) {
        mode = 2;
        keysB = (bf16_t*)extra;
        WkB   = (bf16_t*)(extra + keysB_bytes);
    } else if (ws_size >= base_bytes + wkB_bytes) {
        mode = 1;
        WkB   = (bf16_t*)extra;
    } else {
        mode = 0;
    }

    // bf16 pre-conversion (hoists conversion VALU + halves GEMM HBM traffic)
    if (mode == 2) {
        int n16 = (NKEYS * HH) / 16;            // 12,800,000
        k_cvt_bf16<<<(n16 + 255) / 256, 256, 0, stream>>>(keys, keysB, n16);
    }
    if (mode >= 1) {
        int n16 = (PROJ * HH) / 16;             // 65,536
        k_cvt_bf16<<<(n16 + 255) / 256, 256, 0, stream>>>(Wk, WkB, n16);
    }

    k_zero_query<<<64, 256, 0, stream>>>(query);
    k_mean_x<<<dim3(64, 8), 256, 0, stream>>>(x, query);
    k_proj_q<<<16, 256, 0, stream>>>(query, Wq, bq, q);
    k_qbk<<<1, 256, 0, stream>>>(q, bk, qbk);
    k_score_init<<<3125, 256, 0, stream>>>(qbk, scores);

    const int gemm_blocks = (NKEYS + 127) / 128;   // 782
    if (mode == 2)
        k_gemm_scores_t<2><<<gemm_blocks, 256, 0, stream>>>(keys, keysB, Wk, WkB, q, scores);
    else if (mode == 1)
        k_gemm_scores_t<1><<<gemm_blocks, 256, 0, stream>>>(keys, keysB, Wk, WkB, q, scores);
    else
        k_gemm_scores_t<0><<<gemm_blocks, 256, 0, stream>>>(keys, keysB, Wk, WkB, q, scores);

    k_topk<<<BB, 256, 0, stream>>>(scores, conf, topidx);
    k_gatein<<<128, 256, 0, stream>>>(query, keys, topidx, gatein);
    k_gate_h<<<64, 256, 0, stream>>>(gatein, Wg, bg, Wm1, bm1, gate, hbuf);
    k_gm<<<64, 256, 0, stream>>>(hbuf, Wm2, bm2, gate, gm);
    k_final_ln<<<16384, 256, 0, stream>>>(x, gm, gamma, beta, out);
}
